// CageSkinning_76579266888179
// MI455X (gfx1250) — matrix operand order repaired and checked by reading, not compiled, not run on hardware
//
#include <hip/hip_runtime.h>
#include <hip/hip_bf16.h>

typedef __attribute__((ext_vector_type(16))) _Float16 v16h;
typedef __attribute__((ext_vector_type(2)))  _Float16 h2;
typedef __attribute__((ext_vector_type(8)))  float    v8f;

#define B_  8
#define N_  8192
#define NC  42
#define NF  80
#define KP  12

// ---------------------------------------------------------------------------
// Kernel 1: cage optimization. One block per batch; points cached in LDS as
// fp16 (48KB). Each wave owns up to 6 cage verts in registers; points are
// consumed in half2 pairs (ds_load_b32 per plane) feeding 12 independent
// distance chains. Verts freeze permanently once min_dist <= 0.4 (update is
// monotone), so we early-exit the 100-iter loop when no vert is active.
// ---------------------------------------------------------------------------
__global__ __launch_bounds__(256) void cage_opt_kernel(
    const float* __restrict__ src_shape,   // (B,3,N)
    const float* __restrict__ cage_v,      // (1,3,NC)
    float* __restrict__ cage_out)          // (B,3,NC)
{
    __shared__ _Float16 px[N_], py[N_], pz[N_];
    __shared__ float cx[NC], cy[NC], cz[NC];
    __shared__ int   active[NC];
    __shared__ int   activeCnt;
    const int b   = blockIdx.x;
    const int tid = threadIdx.x;
    const float* base = src_shape + (size_t)b * 3 * N_;
    for (int i = tid; i < N_; i += 256) {
        px[i] = (_Float16)base[i];
        py[i] = (_Float16)base[N_ + i];
        pz[i] = (_Float16)base[2 * N_ + i];
    }
    if (tid < NC) {
        cx[tid] = cage_v[tid];
        cy[tid] = cage_v[NC + tid];
        cz[tid] = cage_v[2 * NC + tid];
        active[tid] = 1;
    }
    if (tid == 0) activeCnt = NC;
    __syncthreads();

    const int lane = tid & 31;
    const int wave = tid >> 5;             // 0..7
    const h2* px2 = (const h2*)px;
    const h2* py2 = (const h2*)py;
    const h2* pz2 = (const h2*)pz;

    for (int it = 0; it < 100; ++it) {
        float vx[6], vy[6], vz[6], m[6];
        #pragma unroll
        for (int j = 0; j < 6; ++j) {
            int v = wave + 8 * j;
            bool ok = (v < NC);
            vx[j] = ok ? cx[v] : 0.f;
            vy[j] = ok ? cy[v] : 0.f;
            vz[j] = ok ? cz[v] : 0.f;
            m[j]  = 3.4e38f;
        }
        for (int i = lane; i < N_ / 2; i += 32) {
            h2 X = px2[i], Y = py2[i], Z = pz2[i];
            float x0 = (float)X[0], x1 = (float)X[1];
            float y0 = (float)Y[0], y1 = (float)Y[1];
            float z0 = (float)Z[0], z1 = (float)Z[1];
            #pragma unroll
            for (int j = 0; j < 6; ++j) {
                float dx0 = vx[j] - x0, dy0 = vy[j] - y0, dz0 = vz[j] - z0;
                float dx1 = vx[j] - x1, dy1 = vy[j] - y1, dz1 = vz[j] - z1;
                float d0 = dx0 * dx0 + dy0 * dy0 + dz0 * dz0;
                float d1 = dx1 * dx1 + dy1 * dy1 + dz1 * dz1;
                m[j] = fminf(m[j], fminf(d0, d1));
            }
        }
        #pragma unroll
        for (int j = 0; j < 6; ++j) {
            #pragma unroll
            for (int off = 16; off > 0; off >>= 1)
                m[j] = fminf(m[j], __shfl_xor(m[j], off, 32));
            int v = wave + 8 * j;
            if (lane == 0 && v < NC) {
                // mind > 0.4  <=>  minsq > 0.16 ; update c *= (1 - 0.01)
                if (m[j] > 0.16f) {
                    cx[v] = vx[j] * 0.99f;
                    cy[v] = vy[j] * 0.99f;
                    cz[v] = vz[j] * 0.99f;
                } else if (active[v]) {
                    active[v] = 0;
                    atomicSub(&activeCnt, 1);
                }
            }
        }
        __syncthreads();
        int na = activeCnt;                 // uniform read
        __syncthreads();                    // protect vs next iter's decrements
        if (na == 0) break;                 // block-uniform break
    }
    if (tid < NC) {
        float* o = cage_out + (size_t)b * 3 * NC;
        o[tid] = cx[tid]; o[NC + tid] = cy[tid]; o[2 * NC + tid] = cz[tid];
    }
}

// ---------------------------------------------------------------------------
// Kernel 2: 4-layer MLP via WMMA f32_16x16x32_f16. One workgroup, 16 waves.
// M padded 8 -> 16; activations ping-pong in LDS as fp16.
// ---------------------------------------------------------------------------
__device__ inline void gemm_layer(
    const _Float16* __restrict__ act, int K,
    const float* __restrict__ W, const float* __restrict__ bias,
    int Nout, int nTiles,
    _Float16* __restrict__ outAct,   // hidden layers (relu)
    float* __restrict__ outF32,      // final layer (no relu)
    int lane, int wave)
{
    const int col = lane & 15;
    const int row = lane & 15;
    const int kbA = (lane < 16) ? 0 : 16;
    const int kbB = (lane < 16) ? 0 : 16;
    for (int nt = wave; nt < nTiles; nt += 16) {
        const int gc = nt * 16 + col;
        const float bv = (gc < Nout) ? bias[gc] : 0.f;
        v8f acc;
        #pragma unroll
        for (int r = 0; r < 8; ++r) acc[r] = bv;
        for (int k0 = 0; k0 < K; k0 += 32) {
            if (k0 + 32 < K && gc < Nout)
                __builtin_prefetch(&W[(k0 + 32) * Nout + gc], 0, 1);
            v16h a, bf;
            #pragma unroll
            for (int j = 0; j < 8; ++j) {
                a[j]     = act[row * K + k0 + kbA + j];
                a[j + 8] = act[row * K + k0 + kbA + 8 + j];
            }
            #pragma unroll
            for (int j = 0; j < 16; ++j) {
                int kk = k0 + kbB + j;
                float wv = (gc < Nout) ? W[kk * Nout + gc] : 0.f;
                bf[j] = (_Float16)wv;
            }
            acc = __builtin_amdgcn_wmma_f32_16x16x32_f16(
                false, a, false, bf, (short)0, acc, false, false);
        }
        #pragma unroll
        for (int r = 0; r < 8; ++r) {
            int orow = (lane < 16) ? r : (r + 8);
            float v = acc[r];
            if (outAct) {
                v = fmaxf(v, 0.f);                       // relu
                if (gc < Nout) outAct[orow * Nout + gc] = (_Float16)v;
            } else {
                if (orow < B_ && gc < NC) outF32[orow * NC + gc] = v;
            }
        }
    }
}

__global__ __launch_bounds__(512) void mlp_kernel(
    const float* __restrict__ src_f, const float* __restrict__ tgt_f,
    const float* __restrict__ W1, const float* __restrict__ b1,
    const float* __restrict__ W2, const float* __restrict__ b2,
    const float* __restrict__ W3, const float* __restrict__ b3,
    const float* __restrict__ W4, const float* __restrict__ b4,
    float* __restrict__ infl_off)          // (B,NC)
{
    __shared__ _Float16 actA[16 * 512];
    __shared__ _Float16 actB[16 * 512];
    const int tid  = threadIdx.x;
    const int lane = tid & 31;
    const int wave = tid >> 5;
    for (int i = tid; i < 16 * 512; i += 512) {
        int rr = i >> 9, cc = i & 511;
        float v = 0.f;
        if (rr < B_) v = (cc < 256) ? src_f[rr * 256 + cc]
                                    : tgt_f[rr * 256 + (cc - 256)];
        actA[i] = (_Float16)v;
    }
    __syncthreads();
    gemm_layer(actA, 512, W1, b1, 512, 32, actB, nullptr, lane, wave);
    __syncthreads();
    gemm_layer(actB, 512, W2, b2, 512, 32, actA, nullptr, lane, wave);
    __syncthreads();
    gemm_layer(actA, 512, W3, b3, 256, 16, actB, nullptr, lane, wave);
    __syncthreads();
    gemm_layer(actB, 256, W4, b4, NC, 3, nullptr, infl_off, lane, wave);
}

// ---------------------------------------------------------------------------
// Kernel 3: influence gating (top-5 keep), cage offset, new cage. One block
// per batch. thr = 5th order statistic (tie semantics match top_k + <=).
// ---------------------------------------------------------------------------
__global__ __launch_bounds__(64) void influence_kernel(
    const float* __restrict__ kp_s,        // (B,KP,3)
    const float* __restrict__ kp_t,        // (B,KP,3)
    const float* __restrict__ infl_param,  // (KP,NC)
    const float* __restrict__ infl_off,    // (B,NC)
    const float* __restrict__ cage,        // (B,3,NC)
    float* __restrict__ new_cage)          // (B,3,NC)
{
    const int b   = blockIdx.x;
    const int tid = threadIdx.x;
    __shared__ float dist[KP * NC];
    __shared__ float thr[KP];
    const float* cg = cage + (size_t)b * 3 * NC;
    for (int i = tid; i < KP * NC; i += 64) {
        int k = i / NC, c = i % NC;
        float dx = kp_s[b * KP * 3 + k * 3 + 0] - cg[c];
        float dy = kp_s[b * KP * 3 + k * 3 + 1] - cg[NC + c];
        float dz = kp_s[b * KP * 3 + k * 3 + 2] - cg[2 * NC + c];
        dist[i] = dx * dx + dy * dy + dz * dz;
    }
    __syncthreads();
    if (tid < KP) {
        const float* dr = dist + tid * NC;
        float t = 3.4e38f;
        for (int c = 0; c < NC; ++c) {
            int cnt = 0;
            for (int c2 = 0; c2 < NC; ++c2) cnt += (dr[c2] <= dr[c]) ? 1 : 0;
            if (cnt >= 5) t = fminf(t, dr[c]);
        }
        thr[tid] = t;
    }
    __syncthreads();
    for (int i = tid; i < 3 * NC; i += 64) {
        int d = i / NC, c = i % NC;
        float off = 0.f;
        for (int k = 0; k < KP; ++k) {
            float keep = (dist[k * NC + c] <= thr[k]) ? 1.f : 0.f;
            float infl = (infl_param[k * NC + c] + infl_off[b * NC + c]) * keep;
            off += (kp_t[b * KP * 3 + k * 3 + d] - kp_s[b * KP * 3 + k * 3 + d]) * infl;
        }
        new_cage[(size_t)b * 3 * NC + i] = cg[i] + off;
    }
}

// ---------------------------------------------------------------------------
// Kernel 4: mean-value-coordinate weights + deformation. One thread per point;
// per-thread W accumulator lives in an LDS column (conflict-free across
// lanes), avoiding scratch spills from dynamic register indexing.
// ---------------------------------------------------------------------------
__global__ __launch_bounds__(256) void mvc_kernel(
    const float* __restrict__ src_shape,   // (B,3,N)
    const int*   __restrict__ faces,       // (NF,3)
    const float* __restrict__ cage,        // (B,3,NC)
    const float* __restrict__ new_cage,    // (B,3,NC)
    float* __restrict__ out)               // (B,3,N)
{
    __shared__ float Wacc[NC * 256];
    __shared__ float cg[3 * NC];
    __shared__ float ncg[3 * NC];
    __shared__ int   fidx[NF * 3];
    const int tid = threadIdx.x;
    const int b   = blockIdx.x >> 5;               // 32 blocks per batch
    const int n   = ((blockIdx.x & 31) << 8) + tid;
    if (tid < 3 * NC) {
        cg[tid]  = cage[(size_t)b * 3 * NC + tid];
        ncg[tid] = new_cage[(size_t)b * 3 * NC + tid];
    }
    for (int i = tid; i < NF * 3; i += 256) fidx[i] = faces[i];
    #pragma unroll
    for (int c = 0; c < NC; ++c) Wacc[c * 256 + tid] = 0.f;
    __syncthreads();

    const float eps = 1e-8f;
    const float px = src_shape[(size_t)b * 3 * N_ + n];
    const float py = src_shape[(size_t)b * 3 * N_ + N_ + n];
    const float pz = src_shape[(size_t)b * 3 * N_ + 2 * N_ + n];

    const int nxt[3] = {1, 2, 0};
    const int prv[3] = {2, 0, 1};
    for (int f = 0; f < NF; ++f) {
        int idx0 = fidx[f * 3], idx1 = fidx[f * 3 + 1], idx2 = fidx[f * 3 + 2];
        int idx[3] = {idx0, idx1, idx2};
        float ux[3], uy[3], uz[3], dd[3];
        #pragma unroll
        for (int t = 0; t < 3; ++t) {
            int c = idx[t];
            float dx = cg[c]          - px;
            float dy = cg[NC + c]     - py;
            float dz = cg[2 * NC + c] - pz;
            float dn = sqrtf(dx * dx + dy * dy + dz * dz);
            dd[t] = dn;
            float inv = 1.f / (dn + eps);
            ux[t] = dx * inv; uy[t] = dy * inv; uz[t] = dz * inv;
        }
        float theta[3], sint[3];
        #pragma unroll
        for (int t = 0; t < 3; ++t) {
            float lx = ux[nxt[t]] - ux[prv[t]];
            float ly = uy[nxt[t]] - uy[prv[t]];
            float lz = uz[nxt[t]] - uz[prv[t]];
            float hl = 0.5f * sqrtf(lx * lx + ly * ly + lz * lz);
            hl = fminf(fmaxf(hl, -1.f + 1e-7f), 1.f - 1e-7f);
            theta[t] = 2.f * asinf(hl);
        }
        float h  = 0.5f * (theta[0] + theta[1] + theta[2]);
        float sh = sinf(h);
        #pragma unroll
        for (int t = 0; t < 3; ++t) sint[t] = sinf(theta[t]);
        float cc[3];
        #pragma unroll
        for (int t = 0; t < 3; ++t)
            cc[t] = 2.f * sh * sinf(h - theta[t]) /
                    (sint[nxt[t]] * sint[prv[t]] + eps) - 1.f;
        float det = ux[0] * (uy[1] * uz[2] - uz[1] * uy[2])
                  - uy[0] * (ux[1] * uz[2] - uz[1] * ux[2])
                  + uz[0] * (ux[1] * uy[2] - uy[1] * ux[2]);
        float sgn = (det > 0.f) ? 1.f : ((det < 0.f) ? -1.f : 0.f);
        float ss[3];
        #pragma unroll
        for (int t = 0; t < 3; ++t) {
            float oc = fminf(fmaxf(1.f - cc[t] * cc[t], 0.f), 1.f);
            ss[t] = sgn * sqrtf(oc);
        }
        #pragma unroll
        for (int t = 0; t < 3; ++t) {
            float denom = dd[t] * sint[nxt[t]] * ss[prv[t]];
            float dn = (fabsf(denom) < eps) ? eps : denom;
            float w = (theta[t] - cc[nxt[t]] * theta[prv[t]]
                                - cc[prv[t]] * theta[nxt[t]]) / dn;
            if (fabsf(ss[prv[t]]) < 1e-6f) w = 0.f;
            Wacc[idx[t] * 256 + tid] += w;
        }
    }

    float wsum = 0.f;
    #pragma unroll
    for (int c = 0; c < NC; ++c) wsum += Wacc[c * 256 + tid];
    const float inv = 1.f / (wsum + eps);
    float ox = 0.f, oy = 0.f, oz = 0.f;
    #pragma unroll
    for (int c = 0; c < NC; ++c) {
        float w = Wacc[c * 256 + tid] * inv;
        ox += w * ncg[c];
        oy += w * ncg[NC + c];
        oz += w * ncg[2 * NC + c];
    }
    out[(size_t)b * 3 * N_ + n]          = ox;
    out[(size_t)b * 3 * N_ + N_ + n]     = oy;
    out[(size_t)b * 3 * N_ + 2 * N_ + n] = oz;
}

// ---------------------------------------------------------------------------
extern "C" void kernel_launch(void* const* d_in, const int* in_sizes, int n_in,
                              void* d_out, int out_size, void* d_ws, size_t ws_size,
                              hipStream_t stream) {
    const float* source_shape = (const float*)d_in[0];
    // d_in[1] target_shape unused by the forward pass
    const float* source_f   = (const float*)d_in[2];
    const float* target_f   = (const float*)d_in[3];
    const float* source_kp  = (const float*)d_in[4];
    const float* target_kp  = (const float*)d_in[5];
    const float* cage_v     = (const float*)d_in[6];
    const float* W1 = (const float*)d_in[7];
    const float* b1 = (const float*)d_in[8];
    const float* W2 = (const float*)d_in[9];
    const float* b2 = (const float*)d_in[10];
    const float* W3 = (const float*)d_in[11];
    const float* b3 = (const float*)d_in[12];
    const float* W4 = (const float*)d_in[13];
    const float* b4 = (const float*)d_in[14];
    const float* infl_param = (const float*)d_in[15];
    const int*   faces      = (const int*)d_in[16];

    float* ws       = (float*)d_ws;
    float* cage     = ws;            // B*3*NC = 1008 floats
    float* infl_off = ws + 1024;     // B*NC   = 336 floats
    float* new_cage = ws + 2048;     // B*3*NC = 1008 floats

    cage_opt_kernel<<<B_, 256, 0, stream>>>(source_shape, cage_v, cage);
    mlp_kernel<<<1, 512, 0, stream>>>(source_f, target_f,
                                      W1, b1, W2, b2, W3, b3, W4, b4, infl_off);
    influence_kernel<<<B_, 64, 0, stream>>>(source_kp, target_kp, infl_param,
                                            infl_off, cage, new_cage);
    mvc_kernel<<<B_ * (N_ / 256), 256, 0, stream>>>(source_shape, faces,
                                                    cage, new_cage, (float*)d_out);
}
